// CRF_24369644437666
// MI455X (gfx1250) — matrix operand order repaired and checked by reading, not compiled, not run on hardware
//
#include <hip/hip_runtime.h>
#include <math.h>
#include <stdint.h>

// Problem constants (from reference)
#define BB 128
#define SS 512
#define TT 256
#define RB 16   // batches per workgroup (WMMA M)

typedef __attribute__((ext_vector_type(16))) _Float16 v16h;
typedef __attribute__((ext_vector_type(8)))  _Float16 v8h;
typedef __attribute__((ext_vector_type(8)))  float    v8f;

// ---- LDS carve (bytes). Emission double-buffer aliases the setup-only
//      exp(transitions) staging area. ----
#define EB_BYTES   (TT * TT * 2 * 2)            // 262144 : EBhi + EBlo (setup only)
#define OFF_EMIT0  0                            // 16384  : aliases EB region
#define OFF_EMIT1  (RB * TT * 4)                // 16384  : aliases EB region
#define OFF_ALPHA  EB_BYTES                     // 16384
#define OFF_PHI    (OFF_ALPHA + RB * TT * 4)    // 8192
#define OFF_PLO    (OFF_PHI + RB * TT * 2)      // 8192
#define OFF_RED    (OFF_PLO + RB * TT * 2)      // 1024
#define OFF_AMAX   (OFF_RED + RB * 16 * 4)      // 64
#define OFF_MASK   (OFF_AMAX + 64)              // 8192 : mask[16][512] bytes
#define SMEM_TOTAL (OFF_MASK + RB * SS)         // 304192 bytes (< 320KB/WGP)

// ---------------------------------------------------------------------------
// Kernel 1: gold (numerator) score per batch.  Cheap gather scan.
// ---------------------------------------------------------------------------
__global__ void crf_gold_kernel(const float* __restrict__ emissions,
                                const int* __restrict__ tags,
                                const unsigned char* __restrict__ mask,
                                const float* __restrict__ transitions,
                                const float* __restrict__ start_t,
                                const float* __restrict__ end_t,
                                float* __restrict__ out_num)
{
    int b = blockIdx.x * blockDim.x + threadIdx.x;
    if (b >= BB) return;
    const int* tg = tags + (size_t)b * SS;
    const unsigned char* mk = mask + (size_t)b * SS;

    int prev = tg[0];
    float score = start_t[prev] + emissions[((size_t)b * SS) * TT + prev];
    int cnt = mk[0] ? 1 : 0;
    for (int t = 1; t < SS; ++t) {
        int cur = tg[t];
        if (mk[t]) {
            score += emissions[((size_t)b * SS + t) * TT + cur]
                   + transitions[(size_t)prev * TT + cur];
            cnt++;
        }
        prev = cur;
    }
    if (cnt < 1) cnt = 1;
    int last = tg[cnt - 1];
    out_num[b] = score + end_t[last];
}

// ---------------------------------------------------------------------------
// Async global->LDS prefetch of one 64-byte chunk group (4 x b128).
// INST_OFFSET is added on both the LDS and global side (ISA 10.x), so one
// lds-address + one global-offset VGPR covers 64 contiguous bytes.
// ---------------------------------------------------------------------------
__device__ __forceinline__ void async_copy64(uint32_t lds_addr, uint32_t gbyte_off,
                                             const float* base)
{
    asm volatile(
        "global_load_async_to_lds_b128 %0, %1, %2\n\t"
        "global_load_async_to_lds_b128 %0, %1, %2 offset:16\n\t"
        "global_load_async_to_lds_b128 %0, %1, %2 offset:32\n\t"
        "global_load_async_to_lds_b128 %0, %1, %2 offset:48"
        :
        : "v"(lds_addr), "v"(gbyte_off), "s"(base)
        : "memory");
}

// ---------------------------------------------------------------------------
// Kernel 2: forward algorithm (log partition), WMMA matmul core.
// One workgroup = 16 batches, 8 waves.  Each wave owns 2 column tiles (32
// tags) and holds its B-matrix (split-f16 exp(trans)^T) in registers for the
// whole recursion.  Emissions double-buffered via async-to-LDS prefetch.
// ---------------------------------------------------------------------------
__global__ __launch_bounds__(256, 1)
void crf_forward_kernel(const float* __restrict__ emissions,
                        const unsigned char* __restrict__ mask,
                        const float* __restrict__ transitions,
                        const float* __restrict__ start_t,
                        const float* __restrict__ end_t,
                        float* __restrict__ out_den)
{
    extern __shared__ char smem[];
    _Float16* EBhi  = (_Float16*)(smem);               // setup only
    _Float16* EBlo  = EBhi + TT * TT;                  // setup only
    float*    alpha = (float*)(smem + OFF_ALPHA);      // [RB][TT]
    _Float16* Phi   = (_Float16*)(smem + OFF_PHI);     // [RB][TT]
    _Float16* Plo   = (_Float16*)(smem + OFF_PLO);     // [RB][TT]
    float*    red   = (float*)(smem + OFF_RED);        // [RB][16]
    float*    amax  = (float*)(smem + OFF_AMAX);       // [RB]
    unsigned char* maskA = (unsigned char*)(smem + OFF_MASK);   // [RB][SS]

    const int tid  = threadIdx.x;
    const int lane = tid & 31;
    const int wave = tid >> 5;
    const int b0   = blockIdx.x * RB;

    const int row16 = tid >> 4;   // 0..15 : batch row in prep phases
    const int l16   = tid & 15;

    // LDS raw base of dynamic shared memory (static size is 0 here)
    const uint32_t lds0 = __builtin_amdgcn_groupstaticsize();
    // this thread's emission chunk (row16, 16 floats at col l16*16)
    const uint32_t emit_lds_off = lds0 + (uint32_t)((row16 * TT + l16 * 16) * 4);
    const uint32_t emit_goff_base =
        (uint32_t)(((b0 + row16) * SS) * TT + l16 * 16) * 4u;

    // --- Setup 1: E^T = exp(transitions)^T, split into f16 hi/lo, into LDS ---
    for (int idx = tid; idx < TT * TT; idx += 256) {
        int k = idx >> 8;          // source row of transitions
        int n = idx & 255;         // source col (consecutive -> coalesced read)
        float e = expf(transitions[idx]);
        _Float16 h = (_Float16)e;
        EBhi[n * TT + k] = h;
        EBlo[n * TT + k] = (_Float16)(e - (float)h);
    }

    // --- Setup 2: alpha[b][j] = start[j] + emissions[b][0][j]; mask stage ---
    for (int j = 0; j < 16; ++j) {
        int col = l16 * 16 + j;
        alpha[row16 * TT + col] =
            start_t[col] + emissions[((size_t)(b0 + row16) * SS) * TT + col];
    }
    {
        const uint32_t* gm = (const uint32_t*)(mask + (size_t)b0 * SS);
        uint32_t* lm = (uint32_t*)maskA;
        for (int i = tid; i < RB * SS / 4; i += 256) lm[i] = gm[i];
    }
    __syncthreads();

    // --- Setup 3: load this wave's B tiles (2 N-tiles x 8 K-tiles, hi+lo) ---
    // f16 B(32x16) in this code's k order (A matches): lanes 0-15 hold K=0..15 of column N=lane,
    //               lanes 16-31 hold K=16..31 of column N=lane-16.
    const int nt0    = wave * 2;
    const int colL   = lane & 15;
    const int hiHalf = lane >> 4;
    v16h bhi[2][8], blo[2][8];
#pragma unroll
    for (int nn = 0; nn < 2; ++nn) {
        int n = (nt0 + nn) * 16 + colL;
#pragma unroll
        for (int kt = 0; kt < 8; ++kt) {
            int kb = kt * 32 + hiHalf * 16;
            const v8h* ph = (const v8h*)(EBhi + n * TT + kb);
            v16h th; ((v8h*)&th)[0] = ph[0]; ((v8h*)&th)[1] = ph[1];
            bhi[nn][kt] = th;
            const v8h* pl = (const v8h*)(EBlo + n * TT + kb);
            v16h tl; ((v8h*)&tl)[0] = pl[0]; ((v8h*)&tl)[1] = pl[1];
            blo[nn][kt] = tl;
        }
    }
    // All EB reads done before async writes alias the region:
    __syncthreads();

    // Prime the pipeline: prefetch emissions for t = 1 into buffer 1.
    async_copy64(emit_lds_off + OFF_EMIT1, emit_goff_base + (uint32_t)(1 * TT * 4),
                 emissions);

    // f16 A(16x32) in this code's k order (B matches): lanes 0-15 (row M=lane) hold K 0..15,
    //               lanes 16-31 (row M=lane-16) hold K 16..31.
    const int aoff  = (lane < 16) ? 0 : 16;
    const int arow  = lane & 15;
    const int rbase = (lane >> 4) * 8;   // C/D: VGPR v -> row v (+8 upper lanes)

    // --- Recursion over time steps ---
    for (int t = 1; t < SS; ++t) {
        // Phase A.1: partial row max of alpha
        {
            float pm = -INFINITY;
#pragma unroll
            for (int i = 0; i < 16; ++i)
                pm = fmaxf(pm, alpha[row16 * TT + l16 + i * 16]);
            red[row16 * 16 + l16] = pm;
        }
        __syncthreads();

        // Phase A.2: prefetch t+1, finish max, build split-f16 P
        {
            if (t + 1 < SS) {
                uint32_t buf = ((t + 1) & 1) ? OFF_EMIT1 : OFF_EMIT0;
                async_copy64(emit_lds_off + buf,
                             emit_goff_base + (uint32_t)((t + 1) * TT * 4),
                             emissions);
            }

            float am = -INFINITY;
#pragma unroll
            for (int i = 0; i < 16; ++i)
                am = fmaxf(am, red[row16 * 16 + i]);
            if (l16 == 0) amax[row16] = am;
#pragma unroll
            for (int i = 0; i < 16; ++i) {
                int col = l16 + i * 16;
                float pf = __expf(alpha[row16 * TT + col] - am);
                _Float16 h = (_Float16)pf;
                Phi[row16 * TT + col] = h;
                Plo[row16 * TT + col] = (_Float16)(pf - (float)h);
            }

            // Retire this step's emission prefetch (in-order completion:
            // allow the 4 newest (t+1) to remain outstanding).
            if (t + 1 < SS)
                asm volatile("s_wait_asynccnt 0x4" ::: "memory");
            else
                asm volatile("s_wait_asynccnt 0x0" ::: "memory");
        }
        __syncthreads();

        // Phase B: S = P x E via WMMA (split precision), then alpha update
        {
            const float* emitc =
                (const float*)(smem + ((t & 1) ? OFF_EMIT1 : OFF_EMIT0));

            v8f acc[2];
#pragma unroll
            for (int nn = 0; nn < 2; ++nn)
                acc[nn] = (v8f){0.f, 0.f, 0.f, 0.f, 0.f, 0.f, 0.f, 0.f};

#pragma unroll
            for (int kt = 0; kt < 8; ++kt) {
                int kb = kt * 32;
                v16h ah, al;
                ((v8h*)&ah)[0] = *(const v8h*)(Phi + arow * TT + kb + aoff);
                ((v8h*)&ah)[1] = *(const v8h*)(Phi + arow * TT + kb + 8 + aoff);
                ((v8h*)&al)[0] = *(const v8h*)(Plo + arow * TT + kb + aoff);
                ((v8h*)&al)[1] = *(const v8h*)(Plo + arow * TT + kb + 8 + aoff);
#pragma unroll
                for (int nn = 0; nn < 2; ++nn) {
                    acc[nn] = __builtin_amdgcn_wmma_f32_16x16x32_f16(
                        false, ah, false, bhi[nn][kt], (short)0, acc[nn], false, false);
                    acc[nn] = __builtin_amdgcn_wmma_f32_16x16x32_f16(
                        false, al, false, bhi[nn][kt], (short)0, acc[nn], false, false);
                    acc[nn] = __builtin_amdgcn_wmma_f32_16x16x32_f16(
                        false, ah, false, blo[nn][kt], (short)0, acc[nn], false, false);
                }
            }

#pragma unroll
            for (int nn = 0; nn < 2; ++nn) {
                int col = (nt0 + nn) * 16 + colL;
#pragma unroll
                for (int v = 0; v < 8; ++v) {
                    int r = rbase + v;
                    float s   = acc[nn][v];
                    float na  = __logf(s) + amax[r] + emitc[r * TT + col];
                    float old = alpha[r * TT + col];
                    alpha[r * TT + col] =
                        (maskA[r * SS + t] != 0) ? na : old;
                }
            }
        }
        __syncthreads();
    }

    // --- Final: log_den[b] = logsumexp_j(alpha[b][j] + end_t[j]) ---
    {
        float pm = -INFINITY;
#pragma unroll
        for (int i = 0; i < 16; ++i) {
            int col = l16 + i * 16;
            pm = fmaxf(pm, alpha[row16 * TT + col] + end_t[col]);
        }
        red[row16 * 16 + l16] = pm;
        __syncthreads();
        float am = -INFINITY;
#pragma unroll
        for (int i = 0; i < 16; ++i)
            am = fmaxf(am, red[row16 * 16 + i]);
        float ps = 0.0f;
#pragma unroll
        for (int i = 0; i < 16; ++i) {
            int col = l16 + i * 16;
            ps += expf(alpha[row16 * TT + col] + end_t[col] - am);
        }
        __syncthreads();                 // red reuse hazard
        red[row16 * 16 + l16] = ps;
        __syncthreads();
        if (l16 == 0) {
            float s = 0.0f;
#pragma unroll
            for (int i = 0; i < 16; ++i)
                s += red[row16 * 16 + i];
            out_den[b0 + row16] = logf(s) + am;
        }
    }
}

// ---------------------------------------------------------------------------
// Kernel 3: mean(num - den) -> scalar
// ---------------------------------------------------------------------------
__global__ void crf_reduce_kernel(const float* __restrict__ num,
                                  const float* __restrict__ den,
                                  float* __restrict__ out)
{
    __shared__ float sb[BB];
    int t = threadIdx.x;
    sb[t] = num[t] - den[t];
    __syncthreads();
    for (int s = BB / 2; s > 0; s >>= 1) {
        if (t < s) sb[t] += sb[t + s];
        __syncthreads();
    }
    if (t == 0) out[0] = sb[0] / (float)BB;
}

// ---------------------------------------------------------------------------
extern "C" void kernel_launch(void* const* d_in, const int* in_sizes, int n_in,
                              void* d_out, int out_size, void* d_ws, size_t ws_size,
                              hipStream_t stream)
{
    const float*         emissions   = (const float*)d_in[0];
    const int*           tags        = (const int*)d_in[1];
    const unsigned char* mask        = (const unsigned char*)d_in[2];
    const float*         transitions = (const float*)d_in[3];
    const float*         start_t     = (const float*)d_in[4];
    const float*         end_t       = (const float*)d_in[5];
    float* out    = (float*)d_out;
    float* ws_num = (float*)d_ws;
    float* ws_den = ws_num + BB;

    crf_gold_kernel<<<1, BB, 0, stream>>>(emissions, tags, mask, transitions,
                                          start_t, end_t, ws_num);

    crf_forward_kernel<<<BB / RB, 256, SMEM_TOTAL, stream>>>(
        emissions, mask, transitions, start_t, end_t, ws_den);

    crf_reduce_kernel<<<1, BB, 0, stream>>>(ws_num, ws_den, out);
}
